// GNNLayer_45603962749760
// MI455X (gfx1250) — matrix operand order repaired and checked by reading, not compiled, not run on hardware
//
#include <hip/hip_runtime.h>
#include <hip/hip_bf16.h>

#define BS 4
#define NN 4096
#define HX 128
#define HY 64
#define CAP 96
#define WT_LD 136   // halfs per row of transposed-W LDS tile (bank-conflict-free 16B reads)

typedef _Float16 v16h __attribute__((ext_vector_type(16)));
typedef _Float16 v8h  __attribute__((ext_vector_type(8)));
typedef _Float16 v4h  __attribute__((ext_vector_type(4)));
typedef float    v8f  __attribute__((ext_vector_type(8)));
typedef float    v4f  __attribute__((ext_vector_type(4)));

// ---------------------------------------------------------------------------
// Kernel 1: zero CSC counters; precompute c[b][f] = b_up[f] + sum_q y[b][q]*W_up[128+q][f]
// ---------------------------------------------------------------------------
__global__ __launch_bounds__(256) void k_init(const float* __restrict__ y,
                                              const float* __restrict__ W_up,
                                              const float* __restrict__ b_up,
                                              int* __restrict__ cnt,
                                              float* __restrict__ cvec) {
  int tid = blockIdx.x * blockDim.x + threadIdx.x;
  int stride = gridDim.x * blockDim.x;
  for (int i = tid; i < BS * NN; i += stride) cnt[i] = 0;
  if (tid < BS * HX) {
    int b = tid >> 7, f = tid & (HX - 1);
    float acc = b_up[f];
#pragma unroll 4
    for (int q = 0; q < HY; ++q) acc += y[b * HY + q] * W_up[(HX + q) * HX + f];
    cvec[tid] = acc;
  }
}

// ---------------------------------------------------------------------------
// Kernel 2: single streaming pass over E (536 MB = the roofline floor).
// 16B nontemporal loads = two adjacency entries per load (halved issue rate).
// Builds per-target-column counts (deg-1) and padded CSC index lists.
// ---------------------------------------------------------------------------
__device__ __forceinline__ void record_edge(long long t, int* __restrict__ cnt,
                                            int* __restrict__ idxbuf) {
  int j = (int)(t & (NN - 1));
  long long bi = t >> 12;
  int i = (int)(bi & (NN - 1));
  int b = (int)(bi >> 12);
  int col = (b << 12) + j;
  int pos = atomicAdd(&cnt[col], 1);
  if (pos < CAP) idxbuf[(long long)col * CAP + pos] = i;
}

__global__ __launch_bounds__(256) void k_scan(const float* __restrict__ E,
                                              int* __restrict__ cnt,
                                              int* __restrict__ idxbuf) {
  const long long totalPairs = (long long)BS * NN * NN / 2;  // one v4f = 2 adjacency entries
  const long long stride = (long long)gridDim.x * blockDim.x;
  for (long long q = (long long)blockIdx.x * blockDim.x + threadIdx.x;
       q < totalPairs; q += stride) {
    v4f e4 = __builtin_nontemporal_load((const v4f*)E + q);  // stream-through
    long long t0 = q << 1;
    if (e4[1] != 0.0f) record_edge(t0, cnt, idxbuf);
    if (e4[3] != 0.0f) record_edge(t0 + 1, cnt, idxbuf);
  }
}

// ---------------------------------------------------------------------------
// WMMA B-fragment from transposed-W LDS tile.
// B 32x16 f16: lane<16 holds K=ks+0..7,16..23, lane>=16 holds K=ks+8..15,24..31 of its column.
// ---------------------------------------------------------------------------
__device__ __forceinline__ v16h ldsB(const _Float16* __restrict__ Wt, int col, int koff) {
  const _Float16* p = Wt + col * WT_LD + koff;
  v8h b0 = *(const v8h*)p;
  v8h b1 = *(const v8h*)(p + 16);
  v16h b;
#pragma unroll
  for (int e = 0; e < 8; ++e) { b[e] = b0[e]; b[8 + e] = b1[e]; }
  return b;
}

// ---------------------------------------------------------------------------
// Kernel 3: msg = rsqrt(deg[row]) * (X @ W_gcn), WMMA f16, f32 accum.
// 128 blocks x 8 waves; wave owns 16 rows x 128 cols (8 tiles), K=128 in 4 steps.
// ---------------------------------------------------------------------------
__global__ __launch_bounds__(256) void k_msg_gemm(const float* __restrict__ X,
                                                  const float* __restrict__ Wg,
                                                  const int* __restrict__ cnt,
                                                  float* __restrict__ msg) {
  __shared__ _Float16 Wt[HX * WT_LD];     // W_gcn transposed: Wt[f][d]
  int tid = threadIdx.x;
  for (int i = tid; i < HX * HX; i += 256) {
    int d = i >> 7, f = i & (HX - 1);
    Wt[f * WT_LD + d] = (_Float16)Wg[i];
  }
  __syncthreads();

  int wave = tid >> 5, lane = tid & 31;
  int r0 = blockIdx.x * 128 + wave * 16;
  int colb = lane & 15;
  int m = r0 + colb;
  int khalf = (lane >> 4) << 3;           // A-frag K sub-offset: 0 or 8

  v8f acc[8] = {};
#pragma unroll
  for (int kt = 0; kt < 4; ++kt) {
    int ks = kt * 32;
    const float* ap = X + (size_t)m * HX + ks + khalf;
    float4 x0 = *(const float4*)(ap + 0);
    float4 x1 = *(const float4*)(ap + 4);
    float4 x2 = *(const float4*)(ap + 16);
    float4 x3 = *(const float4*)(ap + 20);
    v16h a;
    a[0]=(_Float16)x0.x; a[1]=(_Float16)x0.y; a[2]=(_Float16)x0.z; a[3]=(_Float16)x0.w;
    a[4]=(_Float16)x1.x; a[5]=(_Float16)x1.y; a[6]=(_Float16)x1.z; a[7]=(_Float16)x1.w;
    a[8]=(_Float16)x2.x; a[9]=(_Float16)x2.y; a[10]=(_Float16)x2.z; a[11]=(_Float16)x2.w;
    a[12]=(_Float16)x3.x; a[13]=(_Float16)x3.y; a[14]=(_Float16)x3.z; a[15]=(_Float16)x3.w;
    int koff = ks + ((lane >> 4) << 3);
#pragma unroll
    for (int t = 0; t < 8; ++t) {
      v16h b = ldsB(Wt, t * 16 + colb, koff);
      acc[t] = __builtin_amdgcn_wmma_f32_16x16x32_f16(false, a, false, b,
                                                      (short)0, acc[t], false, false);
    }
  }

  int rowoff = (lane >> 4) << 3;          // D rows r (lanes 0-15) / r+8 (lanes 16-31)
#pragma unroll
  for (int r = 0; r < 8; ++r) {
    int row = r0 + rowoff + r;
    float s = rsqrtf(1.0f + (float)cnt[row]);   // dinv inline (deg = cnt + self-loop)
#pragma unroll
    for (int t = 0; t < 8; ++t)
      msg[(size_t)row * HX + t * 16 + colb] = acc[t][r] * s;
  }
}

// ---------------------------------------------------------------------------
// Kernel 4: agg[j] = dinv[j]*(msg[j] + sum_{i in col j} msg[i]) + b_gcn, out f16.
// One wave per target column. Indices block-loaded 32-at-a-time (coalesced),
// broadcast via wave32 shuffles; upcoming rows prefetched (global_prefetch_b8).
// msg (8 MB) is L2-resident.
// ---------------------------------------------------------------------------
__global__ __launch_bounds__(256) void k_gather(const float* __restrict__ msg,
                                                const int* __restrict__ cnt,
                                                const int* __restrict__ idxbuf,
                                                const float* __restrict__ b_gcn,
                                                _Float16* __restrict__ aggH) {
  int wave = threadIdx.x >> 5, lane = threadIdx.x & 31;
  int gj = blockIdx.x * 8 + wave;
  int b = gj >> 12;
  int f = lane << 2;
  float4 acc = *(const float4*)(msg + (size_t)gj * HX + f);   // self-loop term
  int raw = cnt[gj];
  int mcnt = raw < CAP ? raw : CAP;
  const int* ip = idxbuf + (size_t)gj * CAP;

  for (int e0 = 0; e0 < mcnt; e0 += 32) {
    int nb = mcnt - e0; if (nb > 32) nb = 32;
    bool active = (e0 + lane) < mcnt;
    int myi = ip[active ? (e0 + lane) : 0];                   // one coalesced 128B load
    if (active) {                                             // prefetch my row (512B)
      const char* pf = (const char*)(msg + ((size_t)((b << 12) | myi)) * HX);
      __builtin_prefetch(pf + 0,   0, 0);
      __builtin_prefetch(pf + 128, 0, 0);
      __builtin_prefetch(pf + 256, 0, 0);
      __builtin_prefetch(pf + 384, 0, 0);
    }
    for (int e = 0; e < nb; ++e) {
      int i = __shfl(myi, e, 32);                             // wave32 broadcast
      float4 v = *(const float4*)(msg + ((size_t)((b << 12) | i)) * HX + f);
      acc.x += v.x; acc.y += v.y; acc.z += v.z; acc.w += v.w;
    }
  }

  float s = rsqrtf(1.0f + (float)raw);                        // dinv[j] inline
  float4 bg = *(const float4*)(b_gcn + f);
  v4h o;
  o[0] = (_Float16)(acc.x * s + bg.x);
  o[1] = (_Float16)(acc.y * s + bg.y);
  o[2] = (_Float16)(acc.z * s + bg.z);
  o[3] = (_Float16)(acc.w * s + bg.w);
  *(v4h*)(aggH + (size_t)gj * HX + f) = o;
}

// ---------------------------------------------------------------------------
// Kernel 5: out = LayerNorm(ReLU(agg @ W_up[:128] + c[b])) * gamma + beta.
// WMMA f16 GEMM; LN fused via wave32 xor-shuffles (row lives in one 16-lane half).
// ---------------------------------------------------------------------------
__global__ __launch_bounds__(256) void k_update(const _Float16* __restrict__ aggH,
                                                const float* __restrict__ W_up,
                                                const float* __restrict__ cvec,
                                                const float* __restrict__ gamma,
                                                const float* __restrict__ beta,
                                                float* __restrict__ out) {
  __shared__ _Float16 Wt[HX * WT_LD];     // W_up rows 0..127 transposed
  int tid = threadIdx.x;
  for (int i = tid; i < HX * HX; i += 256) {
    int d = i >> 7, f = i & (HX - 1);
    Wt[f * WT_LD + d] = (_Float16)W_up[i];
  }
  __syncthreads();

  int wave = tid >> 5, lane = tid & 31;
  int r0 = blockIdx.x * 128 + wave * 16;
  int b = r0 >> 12;                       // 128 | 4096: whole block is one batch
  int colb = lane & 15;
  int khalf = (lane >> 4) << 3;

  v8f acc[8] = {};
#pragma unroll
  for (int kt = 0; kt < 4; ++kt) {
    int ks = kt * 32;
    const _Float16* ap = aggH + (size_t)(r0 + colb) * HX + ks + khalf;
    v8h lo = *(const v8h*)ap;
    v8h hi = *(const v8h*)(ap + 16);
    v16h a;
#pragma unroll
    for (int e = 0; e < 8; ++e) { a[e] = lo[e]; a[8 + e] = hi[e]; }
    int koff = ks + ((lane >> 4) << 3);
#pragma unroll
    for (int t = 0; t < 8; ++t) {
      v16h bb = ldsB(Wt, t * 16 + colb, koff);
      acc[t] = __builtin_amdgcn_wmma_f32_16x16x32_f16(false, a, false, bb,
                                                      (short)0, acc[t], false, false);
    }
  }

  float cb[8], ga[8], be[8];
#pragma unroll
  for (int t = 0; t < 8; ++t) {
    int col = t * 16 + colb;
    cb[t] = cvec[b * HX + col];
    ga[t] = gamma[col];
    be[t] = beta[col];
  }

  int rowoff = (lane >> 4) << 3;
#pragma unroll
  for (int r = 0; r < 8; ++r) {
    float z[8];
    float sum = 0.f, sq = 0.f;
#pragma unroll
    for (int t = 0; t < 8; ++t) {
      float v = acc[t][r] + cb[t];
      v = v > 0.f ? v : 0.f;              // ReLU
      z[t] = v; sum += v; sq += v * v;
    }
    // reduce over the 16-lane half that owns this row (xor masks 1..8 stay in-half)
#pragma unroll
    for (int msk = 1; msk < 16; msk <<= 1) {
      sum += __shfl_xor(sum, msk, 32);
      sq  += __shfl_xor(sq,  msk, 32);
    }
    float mean = sum * (1.0f / HX);
    float var  = sq * (1.0f / HX) - mean * mean;
    float rstd = rsqrtf(var + 1e-5f);
    int row = r0 + rowoff + r;
#pragma unroll
    for (int t = 0; t < 8; ++t)
      out[(size_t)row * HX + t * 16 + colb] = (z[t] - mean) * rstd * ga[t] + be[t];
  }
}

// ---------------------------------------------------------------------------
extern "C" void kernel_launch(void* const* d_in, const int* in_sizes, int n_in,
                              void* d_out, int out_size, void* d_ws, size_t ws_size,
                              hipStream_t stream) {
  (void)in_sizes; (void)n_in; (void)out_size; (void)ws_size;
  const float* X     = (const float*)d_in[0];
  const float* E     = (const float*)d_in[1];
  const float* y     = (const float*)d_in[2];
  const float* W_gcn = (const float*)d_in[3];
  const float* b_gcn = (const float*)d_in[4];
  const float* W_up  = (const float*)d_in[5];
  const float* b_up  = (const float*)d_in[6];
  const float* gamma = (const float*)d_in[7];
  const float* beta  = (const float*)d_in[8];
  float* out = (float*)d_out;

  char* ws = (char*)d_ws;
  int*      cnt    = (int*)(ws);                                      // 64 KB
  float*    cvec   = (float*)(ws + (64u << 10));                      // 2 KB
  float*    msg    = (float*)(ws + (128u << 10));                     // 8 MB
  _Float16* aggH   = (_Float16*)(ws + (128u << 10) + (8u << 20));     // 4 MB
  int*      idxbuf = (int*)(ws + (128u << 10) + (12u << 20));         // 6 MB

  k_init    <<<64,    256, 0, stream>>>(y, W_up, b_up, cnt, cvec);
  k_scan    <<<16384, 256, 0, stream>>>(E, cnt, idxbuf);
  k_msg_gemm<<<128,   256, 0, stream>>>(X, W_gcn, cnt, msg);
  k_gather  <<<2048,  256, 0, stream>>>(msg, cnt, idxbuf, b_gcn, aggH);
  k_update  <<<128,   256, 0, stream>>>(aggH, W_up, cvec, gamma, beta, out);
}
